// VectorQuantizedMemory_30142080483337
// MI455X (gfx1250) — matrix operand order repaired and checked by reading, not compiled, not run on hardware
//
#include <hip/hip_runtime.h>

// ---------------------------------------------------------------------------
// VQ-VAE codebook step for MI455X (gfx1250, wave32, WMMA).
// Distance argmin = fused GEMM (split-bf16 hi/lo -> 3x v_wmma_f32_16x16x32_bf16
// per K=32 slab, ~fp32 accuracy) + per-row running argmin. ||x||^2 dropped.
// 256 threads = 8 wave32 (2/SIMD): wave = (rowTile 0..3) x (codeSubtilePair 0..1),
// two accumulators per wave reuse A fragments across 6 WMMAs (8 LDS dwords/WMMA).
// ---------------------------------------------------------------------------

#define N_TOKEN   8192
#define N_EMBED   512
#define T_TOTAL   16384
#define DECAYF    0.99f
#define OMDECAYF  0.01f
#define EPSV      1e-8f
#define RATIOF    0.5f        // N_TOKEN / T_TOTAL

#define TPB       64          // tokens per block (4 row tiles x 16)
#define THREADS   256         // 8 wave32
#define KCODES    64          // codes staged per chunk (4 subtiles of 16)
#define XPITCH    514         // bf16 elems per LDS row (257 dwords, odd -> bank-friendly)
#define XPITCH_U  257

typedef __attribute__((ext_vector_type(16))) __bf16       v16bf;
typedef __attribute__((ext_vector_type(8)))  float        v8f;
typedef __attribute__((ext_vector_type(8)))  unsigned int v8u;

union BF16x16 { v8u u; v16bf b; };

__device__ __forceinline__ unsigned short f32_to_bf16_rne(float f) {
  unsigned int u = __float_as_uint(f);
  u += 0x7FFFu + ((u >> 16) & 1u);          // round-to-nearest-even
  return (unsigned short)(u >> 16);
}
__device__ __forceinline__ float bf16_bits_to_f32(unsigned short h) {
  return __uint_as_float(((unsigned int)h) << 16);
}

// ---------------------------------------------------------------------------
__global__ void vq_zero_kernel(float* __restrict__ p, int n) {
  int i = blockIdx.x * blockDim.x + threadIdx.x;
  if (i < n) p[i] = 0.0f;
}

// one wave per code: exact f32 ||key_k||^2
__global__ __launch_bounds__(128) void vq_keynorm_kernel(
    const float* __restrict__ key_w, float* __restrict__ keynorm2) {
  const int lane = threadIdx.x & 31;
  const int wave = threadIdx.x >> 5;
  const int code = blockIdx.x * 4 + wave;
  const float* row = key_w + (long)code * N_EMBED;
  float s = 0.0f;
  for (int d = lane * 4; d < N_EMBED; d += 128) {
    float4 v = *reinterpret_cast<const float4*>(row + d);
    s += v.x * v.x + v.y * v.y + v.z * v.z + v.w * v.w;
  }
#pragma unroll
  for (int off = 16; off > 0; off >>= 1) s += __shfl_xor(s, off, 32);
  if (lane == 0) keynorm2[code] = s;
}

// ---------------------------------------------------------------------------
// Main fused kernel: argmin over codes + quantized output + EMA scatter.
__global__ __launch_bounds__(THREADS) void vq_main_kernel(
    const float* __restrict__ x,
    const float* __restrict__ key_w,
    const float* __restrict__ value_w,
    const float* __restrict__ keynorm2,
    float* __restrict__ quant_out,
    float* __restrict__ cnt,
    float* __restrict__ wsum) {
  __shared__ unsigned short xhi[TPB * XPITCH];      // 65.8 KB
  __shared__ unsigned short xlo[TPB * XPITCH];      // 65.8 KB
  __shared__ unsigned short khi[KCODES * XPITCH];   // 65.8 KB
  __shared__ unsigned short klo[KCODES * XPITCH];   // 65.8 KB
  __shared__ float scratch[8 * 2 * 16 * 17];        // per-wave two 16x16 tiles (17.4 KB)
  __shared__ float fmin[8 * 16];
  __shared__ int   fidx[8 * 16];
  __shared__ int   idxbuf[TPB];

  const int  tid  = threadIdx.x;
  const int  lane = tid & 31;
  const int  wave = tid >> 5;
  const int  half = lane >> 4;
  const int  l16  = lane & 15;
  const int  rtile = wave & 3;        // which 16-token row tile
  const int  spair = wave >> 2;       // which pair of 16-code subtiles
  const long tokBase = (long)blockIdx.x * TPB;

  // Phase 1: stage x tile as split bf16 (hi + residual lo) into LDS.
  for (int i = tid; i < TPB * (N_EMBED / 4); i += THREADS) {
    const int row = i >> 7;                // N_EMBED/4 == 128
    const int c4  = (i & 127) << 2;
    const float4 v = *reinterpret_cast<const float4*>(x + (tokBase + row) * N_EMBED + c4);
    const float vs[4] = {v.x, v.y, v.z, v.w};
#pragma unroll
    for (int j = 0; j < 4; ++j) {
      unsigned short h = f32_to_bf16_rne(vs[j]);
      xhi[row * XPITCH + c4 + j] = h;
      xlo[row * XPITCH + c4 + j] = f32_to_bf16_rne(vs[j] - bf16_bits_to_f32(h));
    }
  }

  float minv = 3.4e38f;
  int   mini = 0;

  // Per-lane LDS row pointers (dword/pair units).
  const unsigned int* arH = reinterpret_cast<const unsigned int*>(xhi) + (rtile * 16 + l16) * XPITCH_U;
  const unsigned int* arL = reinterpret_cast<const unsigned int*>(xlo) + (rtile * 16 + l16) * XPITCH_U;
  const unsigned int* brH0 = reinterpret_cast<const unsigned int*>(khi) + ((2 * spair) * 16 + l16) * XPITCH_U;
  const unsigned int* brL0 = reinterpret_cast<const unsigned int*>(klo) + ((2 * spair) * 16 + l16) * XPITCH_U;
  const unsigned int* brH1 = brH0 + 16 * XPITCH_U;
  const unsigned int* brL1 = brL0 + 16 * XPITCH_U;
  float* sc0 = scratch + wave * (2 * 16 * 17);
  float* sc1 = sc0 + 16 * 17;

  // Phase 2: scan all 8192 codes in chunks of 64.
  for (int ch = 0; ch < N_TOKEN / KCODES; ++ch) {
    __syncthreads();
    // cooperative key-chunk staging (f32 -> bf16 hi/lo)
    for (int i = tid; i < KCODES * (N_EMBED / 4); i += THREADS) {
      const int row = i >> 7;
      const int c4  = (i & 127) << 2;
      const float4 v = *reinterpret_cast<const float4*>(
          key_w + ((long)ch * KCODES + row) * N_EMBED + c4);
      const float vs[4] = {v.x, v.y, v.z, v.w};
#pragma unroll
      for (int j = 0; j < 4; ++j) {
        unsigned short h = f32_to_bf16_rne(vs[j]);
        khi[row * XPITCH + c4 + j] = h;
        klo[row * XPITCH + c4 + j] = f32_to_bf16_rne(vs[j] - bf16_bits_to_f32(h));
      }
    }
    __syncthreads();

    // S tiles = x_tile(16x512) . key_subtile(512x16)^T, split hi/lo bf16 WMMA.
    v8f acc0 = {0.f, 0.f, 0.f, 0.f, 0.f, 0.f, 0.f, 0.f};
    v8f acc1 = {0.f, 0.f, 0.f, 0.f, 0.f, 0.f, 0.f, 0.f};
#pragma unroll 2
    for (int kk = 0; kk < N_EMBED / 32; ++kk) {
      const int kb2  = kk * 16;           // pair-unit base of this K=32 slab
      const int aoff = kb2 + 8 * half;    // A halves: K 0..15 / 16..31
      const int boff = kb2 + 8 * half;    // B halves: K 0..15 / 16..31
      BF16x16 Ah, Al, Bh0, Bl0, Bh1, Bl1;
#pragma unroll
      for (int p = 0; p < 4; ++p) {
        Ah.u[p]     = arH[aoff + p];
        Ah.u[p + 4] = arH[aoff + 4 + p];
        Al.u[p]     = arL[aoff + p];
        Al.u[p + 4] = arL[aoff + 4 + p];
      }
#pragma unroll
      for (int p = 0; p < 8; ++p) {
        Bh0.u[p] = brH0[boff + p];
        Bl0.u[p] = brL0[boff + p];
        Bh1.u[p] = brH1[boff + p];
        Bl1.u[p] = brL1[boff + p];
      }
      acc0 = __builtin_amdgcn_wmma_f32_16x16x32_bf16(false, Ah.b, false, Bh0.b, (short)0, acc0, false, false);
      acc0 = __builtin_amdgcn_wmma_f32_16x16x32_bf16(false, Ah.b, false, Bl0.b, (short)0, acc0, false, false);
      acc0 = __builtin_amdgcn_wmma_f32_16x16x32_bf16(false, Al.b, false, Bh0.b, (short)0, acc0, false, false);
      acc1 = __builtin_amdgcn_wmma_f32_16x16x32_bf16(false, Ah.b, false, Bh1.b, (short)0, acc1, false, false);
      acc1 = __builtin_amdgcn_wmma_f32_16x16x32_bf16(false, Ah.b, false, Bl1.b, (short)0, acc1, false, false);
      acc1 = __builtin_amdgcn_wmma_f32_16x16x32_bf16(false, Al.b, false, Bh1.b, (short)0, acc1, false, false);
    }

    // dist(m,n) = ||k_n||^2 - 2*S(m,n)   (||x_m||^2 constant per row -> dropped)
    const int   codeA = ch * KCODES + (2 * spair) * 16;  // subtile 0 base
    const int   codeB = codeA + 16;                      // subtile 1 base
    const float knA   = keynorm2[codeA + l16];
    const float knB   = keynorm2[codeB + l16];
#pragma unroll
    for (int r = 0; r < 8; ++r) {
      sc0[(r + 8 * half) * 17 + l16] = knA - 2.0f * acc0[r];
      sc1[(r + 8 * half) * 17 + l16] = knB - 2.0f * acc1[r];
    }
    __syncthreads();
    if (lane < 16) {  // lane j owns running argmin of token (rtile*16+j) over this wave's codes
      const float* r0 = sc0 + lane * 17;
      const float* r1 = sc1 + lane * 17;
#pragma unroll
      for (int n = 0; n < 16; ++n) {      // ascending code order, strict < => first-min
        float v = r0[n];
        if (v < minv) { minv = v; mini = codeA + n; }
      }
#pragma unroll
      for (int n = 0; n < 16; ++n) {
        float v = r1[n];
        if (v < minv) { minv = v; mini = codeB + n; }
      }
    }
  }

  // Merge the two code-halves per token: lexicographic (value, index) ==
  // global first-minimum, matching jnp.argmin.
  if (lane < 16) {
    fmin[wave * 16 + lane] = minv;
    fidx[wave * 16 + lane] = mini;
  }
  __syncthreads();
  if (wave < 4 && lane < 16) {
    const float v0 = fmin[wave * 16 + lane];
    const float v1 = fmin[(wave + 4) * 16 + lane];
    const int   i0 = fidx[wave * 16 + lane];
    const int   i1 = fidx[(wave + 4) * 16 + lane];
    const int best = (v1 < v0 || (v1 == v0 && i1 < i0)) ? i1 : i0;
    idxbuf[wave * 16 + lane] = best;
    atomicAdd(&cnt[best], RATIOF);       // cnts scatter (ratio-scaled)
  }
  __syncthreads();

  // Phase 3: quantized = x + value_w[idx]; scatter ws += ratio * x.
  for (int i = tid; i < TPB * (N_EMBED / 4); i += THREADS) {
    const int  row  = i >> 7;
    const int  c4   = (i & 127) << 2;
    const long tok  = tokBase + row;
    const int  kidx = idxbuf[row];
    const float4 xv = *reinterpret_cast<const float4*>(x + tok * N_EMBED + c4);
    const float4 vv = *reinterpret_cast<const float4*>(value_w + (long)kidx * N_EMBED + c4);
    float4 q;
    q.x = xv.x + vv.x; q.y = xv.y + vv.y; q.z = xv.z + vv.z; q.w = xv.w + vv.w;
    *reinterpret_cast<float4*>(quant_out + tok * N_EMBED + c4) = q;
    float* wrow = wsum + (long)kidx * N_EMBED + c4;
    atomicAdd(wrow + 0, xv.x * RATIOF);
    atomicAdd(wrow + 1, xv.y * RATIOF);
    atomicAdd(wrow + 2, xv.z * RATIOF);
    atomicAdd(wrow + 3, xv.w * RATIOF);
  }
}

// ---------------------------------------------------------------------------
__global__ __launch_bounds__(256) void vq_ema_n_kernel(
    const float* __restrict__ ema_n, const float* __restrict__ cnt,
    float* __restrict__ ema_n_out, float* __restrict__ nsum) {
  const int k = blockIdx.x * 256 + threadIdx.x;
  const float en = ema_n[k] * DECAYF + OMDECAYF * cnt[k];
  ema_n_out[k] = en;
  atomicAdd(nsum, en);
}

__global__ __launch_bounds__(256) void vq_final_kernel(
    const float* __restrict__ ema_w, const float* __restrict__ wsum,
    const float* __restrict__ ema_n_out, const float* __restrict__ nsum,
    float* __restrict__ ema_w_out, float* __restrict__ key_out) {
  const long i = (long)blockIdx.x * 256 + threadIdx.x;
  const int  k = (int)(i >> 9);  // / N_EMBED
  const float ew = ema_w[i] * DECAYF + OMDECAYF * wsum[i];
  ema_w_out[i] = ew;
  const float ns = *nsum;
  const float en = ema_n_out[k];
  const float sz = (en + EPSV) / (ns + EPSV * (float)N_TOKEN) * ns;
  key_out[i] = ew / sz;
}

// ---------------------------------------------------------------------------
extern "C" void kernel_launch(void* const* d_in, const int* in_sizes, int n_in,
                              void* d_out, int out_size, void* d_ws, size_t ws_size,
                              hipStream_t stream) {
  (void)in_sizes; (void)n_in; (void)out_size; (void)ws_size;

  const float* x       = (const float*)d_in[0];
  const float* key_w   = (const float*)d_in[1];
  const float* value_w = (const float*)d_in[2];
  const float* ema_n   = (const float*)d_in[3];
  const float* ema_w   = (const float*)d_in[4];

  float* out       = (float*)d_out;
  float* quant     = out;                                     // T*D
  float* ema_n_out = out + (size_t)T_TOTAL * N_EMBED;         // +8192
  float* ema_w_out = ema_n_out + N_TOKEN;                     // +N_TOKEN*D
  float* key_out   = ema_w_out + (size_t)N_TOKEN * N_EMBED;

  // workspace carve: [wsum | cnt | nsum | keynorm2]  (~16.9 MB)
  float* wsum     = (float*)d_ws;
  float* cnt      = wsum + (size_t)N_TOKEN * N_EMBED;
  float* nsum     = cnt + N_TOKEN;
  float* keynorm2 = nsum + 1;

  const int zeroN = N_TOKEN * N_EMBED + N_TOKEN + 1;  // wsum+cnt+nsum contiguous
  vq_zero_kernel<<<(zeroN + 255) / 256, 256, 0, stream>>>(wsum, zeroN);
  vq_keynorm_kernel<<<N_TOKEN / 4, 128, 0, stream>>>(key_w, keynorm2);
  vq_main_kernel<<<T_TOTAL / TPB, THREADS, 0, stream>>>(
      x, key_w, value_w, keynorm2, quant, cnt, wsum);
  vq_ema_n_kernel<<<N_TOKEN / 256, 256, 0, stream>>>(ema_n, cnt, ema_n_out, nsum);
  vq_final_kernel<<<(N_TOKEN * N_EMBED) / 256, 256, 0, stream>>>(
      ema_w, wsum, ema_n_out, nsum, ema_w_out, key_out);
}
